// PaiNNInteraction_88897233093048
// MI455X (gfx1250) — matrix operand order repaired and checked by reading, not compiled, not run on hardware
//
#include <hip/hip_runtime.h>
#include <hip/hip_bf16.h>

#define H 256
#define NATOMS 10000
#define NEDGES 320000
#define NRBF 20
#define BM 64

typedef __attribute__((ext_vector_type(16))) __bf16 v16bf;
typedef __attribute__((ext_vector_type(8)))  float  v8f;
typedef __attribute__((ext_vector_type(4)))  float  f4;
typedef __attribute__((ext_vector_type(8)))  unsigned short u16x8;
typedef __attribute__((ext_vector_type(16))) unsigned short u16x16;

union BF16x16 { u16x16 u; v16bf b; };

__device__ __forceinline__ unsigned short f2bf(float f) {
  unsigned u = __builtin_bit_cast(unsigned, f);
  u += 0x7FFFu + ((u >> 16) & 1u);          // round-to-nearest-even
  return (unsigned short)(u >> 16);
}
__device__ __forceinline__ float silu(float x) { return x / (1.0f + __expf(-x)); }

__device__ __forceinline__ v8f wmma_bf16(v16bf a, v16bf b, v8f c) {
  return __builtin_amdgcn_wmma_f32_16x16x32_bf16(false, a, false, b, (short)0, c, false, false);
}

// CDNA5 async DMA: 16B global -> LDS, tracked with ASYNCcnt (no VGPR staging).
__device__ __forceinline__ void async_copy16(unsigned lds_off, unsigned long long gaddr) {
  asm volatile("global_load_async_to_lds_b128 %0, %1, off"
               :: "v"(lds_off), "v"(gaddr) : "memory");
}
__device__ __forceinline__ void wait_async0() {
  asm volatile("s_wait_asynccnt 0x0" ::: "memory");
}

__device__ __forceinline__ void atomic_add_f32(float* p, float v) {
  __hip_atomic_fetch_add(p, v, __ATOMIC_RELAXED, __HIP_MEMORY_SCOPE_AGENT);
}

// A fragment (16x32 bf16, ISA 7.12.2): lanes 0-15: K=0..7 (V0-3), K=16..23 (V4-7);
// lanes 16-31: K=8..15, K=24..31.  Two 16B LDS loads per lane.
__device__ __forceinline__ v16bf afrag(const unsigned short* base, int rbase, int k0,
                                       int stride, int lane) {
  int m = lane & 15, kh = lane >> 4;
  const unsigned short* p = base + (size_t)(rbase + m) * stride + k0 + kh * 8;
  u16x8 lo = *(const u16x8*)p;
  u16x8 hi = *(const u16x8*)(p + 16);
  BF16x16 f;
  f.u = __builtin_shufflevector(lo, hi, 0,1,2,3,4,5,6,7,8,9,10,11,12,13,14,15);
  return f.b;
}

// B fragment from pre-packed weights: one contiguous 32B load per lane (coalesced).
__device__ __forceinline__ v16bf bfrag(const unsigned short* pack, int nt, int kt,
                                       int KT, int lane) {
  const u16x16* tp = (const u16x16*)(pack + ((size_t)(nt * KT + kt) << 9));
  BF16x16 f; f.u = tp[lane];
  return f.b;
}

// -------- prep: LayerNorm + bf16 convert, one wave per row --------
__global__ __launch_bounds__(256)
void ln_bf16(const float* __restrict__ s, const float* __restrict__ g,
             const float* __restrict__ b, unsigned short* __restrict__ out) {
  int row  = blockIdx.x * 8 + (threadIdx.x >> 5);
  int lane = threadIdx.x & 31;
  const float* rp = s + (size_t)row * H + lane * 8;
  f4 a = *(const f4*)rp, c = *(const f4*)(rp + 4);
  float x[8] = {a.x,a.y,a.z,a.w,c.x,c.y,c.z,c.w};
  float sum = 0.f;
#pragma unroll
  for (int i = 0; i < 8; ++i) sum += x[i];
#pragma unroll
  for (int o = 16; o > 0; o >>= 1) sum += __shfl_xor(sum, o, 32);
  float mean = sum * (1.0f / H);
  float vs = 0.f;
#pragma unroll
  for (int i = 0; i < 8; ++i) { float d = x[i] - mean; vs += d * d; }
#pragma unroll
  for (int o = 16; o > 0; o >>= 1) vs += __shfl_xor(vs, o, 32);
  float rinv = rsqrtf(vs * (1.0f / H) + 1e-5f);
  f4 g0 = *(const f4*)(g + lane*8), g1 = *(const f4*)(g + lane*8 + 4);
  f4 b0 = *(const f4*)(b + lane*8), b1 = *(const f4*)(b + lane*8 + 4);
  float gg[8] = {g0.x,g0.y,g0.z,g0.w,g1.x,g1.y,g1.z,g1.w};
  float bb[8] = {b0.x,b0.y,b0.z,b0.w,b1.x,b1.y,b1.z,b1.w};
  u16x8 o8;
#pragma unroll
  for (int i = 0; i < 8; ++i) o8[i] = f2bf((x[i] - mean) * rinv * gg[i] + bb[i]);
  *(u16x8*)(out + (size_t)row * H + lane * 8) = o8;
}

// -------- prep: pack f32 KxN row-major weight into bf16 B-fragment tiles --------
// B layout (matches A): lane n holds column n; lanes 0-15: K=0..7, K=16..23
// (2 per VGPR), lanes 16-31: K=8..15, K=24..31.  Zero-pad K >= Kact (fw1: 20 -> 32).
__global__ __launch_bounds__(256)
void pack_w(const float* __restrict__ w, unsigned short* __restrict__ dst,
            int Kact, int KT, int N) {
  int p = blockIdx.x * 256 + threadIdx.x;
  int total = KT * 32 * N;
  if (p >= total) return;
  int tile = p >> 9, within = p & 511;
  int lane = within >> 4, e = within & 15;
  int r = e >> 1, hi = e & 1;
  int kt = tile % KT, nt = tile / KT;
  int n = nt * 16 + (lane & 15);
  int k = kt * 32 + ((e >> 3) << 4) + ((lane >> 4) << 3) + (e & 7);
  float val = (k < Kact) ? w[(size_t)k * N + n] : 0.0f;
  dst[p] = f2bf(val);
}

// -------- prep: out = concat(s, v); edge kernel accumulates atomically --------
__global__ __launch_bounds__(256)
void init_out(const float* __restrict__ s, const float* __restrict__ v,
              float* __restrict__ out) {
  size_t i = (size_t)blockIdx.x * 256 + threadIdx.x;
  const size_t SE = (size_t)NATOMS * H;
  const size_t TE = SE * 4;
  if (i < SE)      out[i] = s[i];
  else if (i < TE) out[i] = v[i - SE];
}

// -------- fused per-edge kernel: gather + 4 GEMMs + combine + scatter --------
__global__ __launch_bounds__(256)
void painn_edge(const unsigned short* __restrict__ snorm,
                const unsigned short* __restrict__ cw1p,
                const unsigned short* __restrict__ cw2p,
                const unsigned short* __restrict__ fw1p,
                const unsigned short* __restrict__ fw2p,
                const int* __restrict__ idx_i, const int* __restrict__ idx_j,
                const float* __restrict__ rbf, const float* __restrict__ f_cut,
                const float* __restrict__ dir, const float* __restrict__ inv_p,
                const float* __restrict__ cb1, const float* __restrict__ cb2,
                const float* __restrict__ fb1, const float* __restrict__ fb2,
                const float* __restrict__ v_in,
                float* __restrict__ outS, float* __restrict__ outV) {
  extern __shared__ unsigned char smem[];
  unsigned short* sA   = (unsigned short*)smem;   // BM x H  gathered s_normed (bf16)
  unsigned short* sH1  = sA  + BM * H;            // BM x H  silu hidden (context path)
  unsigned short* sG1  = sH1 + BM * H;            // BM x H  silu hidden (filter path)
  unsigned short* sRbf = sG1 + BM * H;            // BM x 32 padded rbf
  int*   sIi = (int*)(sRbf + BM * 32);
  int*   sIj = sIi + BM;
  float* sFc = (float*)(sIj + BM);
  float* sDr = sFc + BM;

  int tid = threadIdx.x, lane = tid & 31, wave = tid >> 5;
  int e0 = blockIdx.x * BM;

  if (tid < BM) {
    sIi[tid] = idx_i[e0 + tid];
    sIj[tid] = idx_j[e0 + tid];
    sFc[tid] = f_cut[e0 + tid] * inv_p[0];        // fold 1/sqrt(neighbors)
  }
  if (tid < BM * 3) sDr[tid] = dir[(size_t)e0 * 3 + tid];
  for (int p = tid; p < BM * 32; p += 256) {
    int r = p >> 5, k = p & 31;
    float val = (k < NRBF) ? rbf[(size_t)(e0 + r) * NRBF + k] : 0.0f;
    sRbf[p] = f2bf(val);
  }
  __builtin_prefetch(cw1p + ((size_t)wave << 12), 0, 0);
  __syncthreads();
  // gather bf16 s_normed rows: async DMA straight into LDS (ASYNCcnt-tracked)
  for (int c = tid; c < BM * 32; c += 256) {     // 64 rows x 32 chunks of 16B
    int r = c >> 5, off = c & 31;
    unsigned lds_off = (unsigned)(unsigned long long)(sA + (size_t)r * H)
                     + (unsigned)(off << 4);
    unsigned long long ga = (unsigned long long)(snorm + (size_t)sIj[r] * H)
                          + (unsigned long long)(off << 4);
    async_copy16(lds_off, ga);
  }
  wait_async0();
  __syncthreads();

  // Phase 1: h1 = silu(sA@cw1 + cb1); g1 = silu(rbf@fw1 + fb1)
#pragma unroll 1
  for (int it = 0; it < 8; ++it) {
    int item = wave + (it << 3);
    int mt = item >> 4, nt = item & 15;
    v8f acc = {};
#pragma unroll
    for (int kt = 0; kt < 8; ++kt)
      acc = wmma_bf16(afrag(sA, mt * 16, kt * 32, H, lane),
                      bfrag(cw1p, nt, kt, 8, lane), acc);
    v8f acc2 = {};
    acc2 = wmma_bf16(afrag(sRbf, mt * 16, 0, 32, lane),
                     bfrag(fw1p, nt, 0, 1, lane), acc2);
    int col = nt * 16 + (lane & 15);
    float b1 = cb1[col], b2 = fb1[col];
    int rb = mt * 16 + ((lane >> 4) << 3);
#pragma unroll
    for (int i = 0; i < 8; ++i) {
      sH1[(size_t)(rb + i) * H + col] = f2bf(silu(acc[i] + b1));
      sG1[(size_t)(rb + i) * H + col] = f2bf(silu(acc2[i] + b2));
    }
  }
  __syncthreads();

  // Phase 2: context/W GEMM pair over all three segments, combine, scatter
#pragma unroll 1
  for (int it = 0; it < 8; ++it) {
    int item = wave + (it << 3);
    int mt = item >> 4, ng = item & 15;
    v8f c0 = {}, c1 = {}, c2 = {}, w0 = {}, w1 = {}, w2 = {};
#pragma unroll 1
    for (int kt = 0; kt < 8; ++kt) {
      v16bf ah = afrag(sH1, mt * 16, kt * 32, H, lane);
      v16bf ag = afrag(sG1, mt * 16, kt * 32, H, lane);
      c0 = wmma_bf16(ah, bfrag(cw2p, ng,      kt, 8, lane), c0);
      c1 = wmma_bf16(ah, bfrag(cw2p, 16 + ng, kt, 8, lane), c1);
      c2 = wmma_bf16(ah, bfrag(cw2p, 32 + ng, kt, 8, lane), c2);
      w0 = wmma_bf16(ag, bfrag(fw2p, ng,      kt, 8, lane), w0);
      w1 = wmma_bf16(ag, bfrag(fw2p, 16 + ng, kt, 8, lane), w1);
      w2 = wmma_bf16(ag, bfrag(fw2p, 32 + ng, kt, 8, lane), w2);
    }
    int col = ng * 16 + (lane & 15);
    float bc0 = cb2[col], bc1 = cb2[col + H], bc2 = cb2[col + 2 * H];
    float bw0 = fb2[col], bw1 = fb2[col + H], bw2 = fb2[col + 2 * H];
    int rb = mt * 16 + ((lane >> 4) << 3);
#pragma unroll
    for (int i = 0; i < 8; ++i) {
      int m = rb + i;
      float sc  = sFc[m];
      float ds  = (c0[i] + bc0) * (w0[i] + bw0) * sc;
      float dvs = (c1[i] + bc1) * (w1[i] + bw1) * sc;
      float dvv = (c2[i] + bc2) * (w2[i] + bw2) * sc;
      int ai = sIi[m], aj = sIj[m];
      atomic_add_f32(&outS[(size_t)ai * H + col], ds);
#pragma unroll
      for (int d = 0; d < 3; ++d) {
        float vj = v_in[((size_t)aj * 3 + d) * H + col];
        atomic_add_f32(&outV[((size_t)ai * 3 + d) * H + col],
                       dvs * sDr[m * 3 + d] + dvv * vj);
      }
    }
  }
}

extern "C" void kernel_launch(void* const* d_in, const int* in_sizes, int n_in,
                              void* d_out, int out_size, void* d_ws, size_t ws_size,
                              hipStream_t stream) {
  (void)in_sizes; (void)n_in; (void)out_size; (void)ws_size;
  const float* s     = (const float*)d_in[0];
  const float* v     = (const float*)d_in[1];
  const int*   idx_i = (const int*)d_in[2];
  const int*   idx_j = (const int*)d_in[3];
  const float* rbf   = (const float*)d_in[4];
  const float* f_cut = (const float*)d_in[5];
  const float* dir   = (const float*)d_in[6];
  const float* invp  = (const float*)d_in[7];
  const float* lng   = (const float*)d_in[8];
  const float* lnb   = (const float*)d_in[9];
  const float* cw1   = (const float*)d_in[10];
  const float* cb1   = (const float*)d_in[11];
  const float* cw2   = (const float*)d_in[12];
  const float* cb2   = (const float*)d_in[13];
  const float* fw1   = (const float*)d_in[14];
  const float* fb1   = (const float*)d_in[15];
  const float* fw2   = (const float*)d_in[16];
  const float* fb2   = (const float*)d_in[17];

  // ws layout (bf16): s_normed 5.12MB | cw1 128KB | cw2 384KB | fw1 16KB | fw2 384KB
  unsigned short* snorm = (unsigned short*)d_ws;
  unsigned short* cw1p  = snorm + (size_t)NATOMS * H;
  unsigned short* cw2p  = cw1p + 256 * 256;
  unsigned short* fw1p  = cw2p + 256 * 768;
  unsigned short* fw2p  = fw1p + 32 * 256;

  float* outS = (float*)d_out;
  float* outV = outS + (size_t)NATOMS * H;

  ln_bf16<<<NATOMS / 8, 256, 0, stream>>>(s, lng, lnb, snorm);
  pack_w<<<(256 * 256 + 255) / 256, 256, 0, stream>>>(cw1, cw1p, 256, 8, 256);
  pack_w<<<(256 * 768 + 255) / 256, 256, 0, stream>>>(cw2, cw2p, 256, 8, 768);
  pack_w<<<(32 * 256 + 255) / 256, 256, 0, stream>>>(fw1, fw1p, 20, 1, 256);
  pack_w<<<(256 * 768 + 255) / 256, 256, 0, stream>>>(fw2, fw2p, 256, 8, 768);
  init_out<<<(NATOMS * H * 4 + 255) / 256, 256, 0, stream>>>(s, v, (float*)d_out);

  size_t smem = (size_t)(BM * H * 3 + BM * 32) * 2 + BM * 8 + BM * 4 + BM * 12;
  painn_edge<<<NEDGES / BM, 256, smem, stream>>>(
      snorm, cw1p, cw2p, fw1p, fw2p, idx_i, idx_j, rbf, f_cut, dir, invp,
      cb1, cb2, fb1, fb2, v, outS, outV);
}
